// SelfAttentionV2_41532333752977
// MI455X (gfx1250) — compile-verified, matrix operand order checked by reading, not run on hardware
//
#include <hip/hip_runtime.h>
#include <stdint.h>

// ---------------------------------------------------------------------------
// Self-attention on gfx1250: Q=xWq+bq, K=xWk+bk, V=xWv+bv;
// out = softmax(QK^T/sqrt(d)) V.   N=4096 tokens, d=1024.
//
// All matmuls: v_wmma_f32_16x16x32_bf16, 64x64 wave tiles (16 WMMA / 16
// ds_load_b128 per wave per k-step). Tile staging: Tensor Data Mover
// (tensor_load_to_lds) with LDS padding, double-buffered, overlapped with
// WMMA compute (TENSORcnt). Wave 0 stages A tiles, wave 1 stages B tiles.
// All B operands are kept/produced in [N,K] row-major form so the TDM can
// stage them with zero in-kernel transposes.
// ---------------------------------------------------------------------------

typedef __attribute__((ext_vector_type(16))) __bf16        bf16x16;
typedef __attribute__((ext_vector_type(8)))  float         floatx8;
typedef __attribute__((ext_vector_type(4)))  unsigned int  u32x4;
typedef __attribute__((ext_vector_type(8)))  int           i32x8;
typedef __attribute__((ext_vector_type(4)))  int           i32x4;

__device__ __forceinline__ unsigned short f32_to_bf16_rne(float f) {
  union { float f; uint32_t u; } x; x.f = f;
  uint32_t r = x.u + 0x7FFFu + ((x.u >> 16) & 1u);
  return (unsigned short)(r >> 16);
}

// ---------------------------------------------------------------------------
// fp32 -> bf16 (grid-stride)
// ---------------------------------------------------------------------------
__global__ __launch_bounds__(256)
void cvt_f32_to_bf16(const float* __restrict__ src,
                     unsigned short* __restrict__ dst, int n) {
  int i = blockIdx.x * blockDim.x + threadIdx.x;
  int stride = gridDim.x * blockDim.x;
  for (; i < n; i += stride) dst[i] = f32_to_bf16_rne(src[i]);
}

// ---------------------------------------------------------------------------
// fp32 [D,D] -> bf16 transposed: dst[n*D+k] = src[k*D+n]; 32x32 LDS tile.
// ---------------------------------------------------------------------------
__global__ __launch_bounds__(256)
void cvt_transpose_f32_to_bf16(const float* __restrict__ src,
                               unsigned short* __restrict__ dst, int D) {
  __shared__ float t[32][33];
  const int bx = blockIdx.x * 32;
  const int by = blockIdx.y * 32;
  const int tx = threadIdx.x & 31;
  const int ty = threadIdx.x >> 5;
  for (int r = ty; r < 32; r += 8)
    t[r][tx] = src[(size_t)(by + r) * D + bx + tx];
  __syncthreads();
  for (int r = ty; r < 32; r += 8)
    dst[(size_t)(bx + r) * D + by + tx] = f32_to_bf16_rne(t[tx][r]);
}

// ---------------------------------------------------------------------------
// TDM: DMA one 2D bf16 tile (tile_d1 rows x tile_d0 elems, row stride
// `stride_elems`) from global into LDS at byte offset `lds_off`, appending
// pad_amount DWORDs per pad_interval stored DWORDs (ISA D# encoding).
// ---------------------------------------------------------------------------
__device__ __forceinline__ void tdm_load_tile_bf16(
    unsigned lds_off, const unsigned short* gsrc,
    unsigned tile_d0, unsigned tile_d1,
    unsigned long long stride_elems,
    unsigned pad_interval, unsigned pad_amount) {
  const unsigned long long ga = (unsigned long long)(uintptr_t)gsrc;

  // D# group 0: count=1 | lds_addr | global_addr[56:0] | type=2
  u32x4 g0;
  g0[0] = 1u;
  g0[1] = lds_off;
  g0[2] = (unsigned)ga;
  g0[3] = (unsigned)((ga >> 32) & 0x01FFFFFFu) | (2u << 30);

  // D# group 1: wg_mask=0 | data_size=1(2B) | pad | tensor dims == tile dims
  // (no clipping) | tile dims | tensor_dim0_stride.
  i32x8 g1;
  g1[0] = (int)((1u << 16) | (1u << 20) |
                (pad_interval << 22) | (pad_amount << 25));
  g1[1] = (int)((tile_d0 & 0xFFFFu) << 16);
  g1[2] = (int)(((tile_d0 >> 16) & 0xFFFFu) | ((tile_d1 & 0xFFFFu) << 16));
  g1[3] = (int)(((tile_d1 >> 16) & 0xFFFFu) | ((tile_d0 & 0xFFFFu) << 16));
  g1[4] = (int)(tile_d1 & 0xFFFFu);
  g1[5] = (int)(unsigned)(stride_elems & 0xFFFFFFFFu);
  g1[6] = (int)(unsigned)((stride_elems >> 32) & 0xFFFFu);
  g1[7] = 0;

  i32x4 z4 = {0, 0, 0, 0};
#if __clang_major__ >= 23
  i32x8 z8 = {0, 0, 0, 0, 0, 0, 0, 0};
  __builtin_amdgcn_tensor_load_to_lds(g0, g1, z4, z4, z8, 0);
#else
  __builtin_amdgcn_tensor_load_to_lds(g0, g1, z4, z4, 0);
#endif
}

// ---------------------------------------------------------------------------
// WMMA GEMM: D = scale * (A @ B^T) + bias
//   A: [M,K] bf16 row-major (lda)
//   B: [N,K] bf16 row-major (ldb)
//   D: [M,N]: fp32 (ldd) or bf16 (ldd); OUT_T stores D^T (bf16, ldd = M dim)
// Block tile 256x128, BK=32, 256 threads = 8 waves (4M x 2N), wave 64x64.
// Per k-step per wave: 16 ds_load_b128 -> 16 WMMAs (A/B fragments each
// reused 4x). TDM double-buffered staging; 40-elem padded LDS rows.
// ---------------------------------------------------------------------------
#define BM 256
#define BN 128
#define BK 32
#define LDSS 40  // 32 data + 4-DWORD pad => 80B rows, 16B aligned

template <bool HAS_BIAS, bool OUT_BF16, bool OUT_T>
__global__ __launch_bounds__(256)
void wmma_gemm(const unsigned short* __restrict__ A,
               const unsigned short* __restrict__ B,
               const float* __restrict__ bias,
               void* __restrict__ Dout,
               int K, int lda, int ldb, int ldd, float scale) {
  __shared__ __align__(16) unsigned short As[2][BM * LDSS];
  __shared__ __align__(16) unsigned short Bs[2][BN * LDSS];

  const int tid    = threadIdx.x;
  const int lane   = tid & 31;
  const int wave   = tid >> 5;
  const int half   = lane >> 4;
  const int lr     = lane & 15;
  const int wave_m = wave >> 1;  // 0..3
  const int wave_n = wave & 1;   // 0..1

  const int bm = blockIdx.y * BM;
  const int bn = blockIdx.x * BN;
  const int nk = K / BK;

  floatx8 acc[4][4] = {};

  // Prologue: stage k-tile 0 into buffer 0 (wave0: A, wave1: B).
  if (wave == 0) {
    tdm_load_tile_bf16((unsigned)(uintptr_t)&As[0][0],
                       A + (size_t)bm * lda, BK, BM, (unsigned long long)lda,
                       /*interval 16DW=*/3, /*pad 4DW=*/3);
  } else if (wave == 1) {
    tdm_load_tile_bf16((unsigned)(uintptr_t)&Bs[0][0],
                       B + (size_t)bn * ldb, BK, BN, (unsigned long long)ldb,
                       3, 3);
  }

  for (int kt = 0; kt < nk; ++kt) {
    const int buf = kt & 1;

    // (a) everyone is done reading buf^1 -> TDM may overwrite it.
    __syncthreads();
    if (wave == 0) {
      if (kt + 1 < nk) {
        tdm_load_tile_bf16((unsigned)(uintptr_t)&As[buf ^ 1][0],
                           A + (size_t)bm * lda + (kt + 1) * BK, BK, BM,
                           (unsigned long long)lda, 3, 3);
        // per-wave TENSORcnt, in-order: <=1 pending => current A tile landed
        __builtin_amdgcn_s_wait_tensorcnt(1);
      } else {
        __builtin_amdgcn_s_wait_tensorcnt(0);
      }
    } else if (wave == 1) {
      if (kt + 1 < nk) {
        tdm_load_tile_bf16((unsigned)(uintptr_t)&Bs[buf ^ 1][0],
                           B + (size_t)bn * ldb + (kt + 1) * BK, BK, BN,
                           (unsigned long long)ldb, 3, 3);
        __builtin_amdgcn_s_wait_tensorcnt(1);
      } else {
        __builtin_amdgcn_s_wait_tensorcnt(0);
      }
    }
    // (b) current buffer visible to all waves.
    __syncthreads();

    // Fragment loads (documented 16-bit operand layout):
    // lane holds row m (A) / row n (B); elem e -> k=(e<8?0:16)+half*8+(e&7)
    union Frag { u32x4 u[2]; bf16x16 v; };
    Frag af[4], bfr[4];
#pragma unroll
    for (int i = 0; i < 4; ++i) {
      const int m = wave_m * 64 + i * 16 + lr;
      const unsigned short* p = &As[buf][m * LDSS + half * 8];
      af[i].u[0] = *(const u32x4*)(p);
      af[i].u[1] = *(const u32x4*)(p + 16);
    }
#pragma unroll
    for (int j = 0; j < 4; ++j) {
      const int n = wave_n * 64 + j * 16 + lr;
      const unsigned short* p = &Bs[buf][n * LDSS + half * 8];
      bfr[j].u[0] = *(const u32x4*)(p);
      bfr[j].u[1] = *(const u32x4*)(p + 16);
    }

#pragma unroll
    for (int i = 0; i < 4; ++i)
#pragma unroll
      for (int j = 0; j < 4; ++j)
        acc[i][j] = __builtin_amdgcn_wmma_f32_16x16x32_bf16(
            false, af[i].v, false, bfr[j].v, (short)0, acc[i][j],
            false, false);
  }

  // Epilogue: D VGPR r -> row = base + half*8 + r, col = base + lr.
#pragma unroll
  for (int i = 0; i < 4; ++i) {
    const int row0 = bm + wave_m * 64 + i * 16 + half * 8;
#pragma unroll
    for (int j = 0; j < 4; ++j) {
      const int col = bn + wave_n * 64 + j * 16 + lr;
      const float bv = HAS_BIAS ? bias[col] : 0.0f;
#pragma unroll
      for (int r = 0; r < 8; ++r) {
        const float val = fmaf(acc[i][j][r], scale, bv);
        if (OUT_T)        // transposed bf16 store: D^T[col][row]
          ((unsigned short*)Dout)[(size_t)col * ldd + row0 + r] =
              f32_to_bf16_rne(val);
        else if (OUT_BF16)
          ((unsigned short*)Dout)[(size_t)(row0 + r) * ldd + col] =
              f32_to_bf16_rne(val);
        else
          ((float*)Dout)[(size_t)(row0 + r) * ldd + col] = val;
      }
    }
  }
}

// ---------------------------------------------------------------------------
// Row softmax: fp32 scores [rows,N] -> bf16 probabilities.
// ---------------------------------------------------------------------------
__global__ __launch_bounds__(256)
void softmax_rows(const float* __restrict__ S,
                  unsigned short* __restrict__ P, int N) {
  __shared__ float red[256];
  const int row = blockIdx.x;
  const int tid = threadIdx.x;
  const float* s = S + (size_t)row * N;
  unsigned short* p = P + (size_t)row * N;

  float lmax = -3.4e38f;
  for (int c = tid; c < N; c += 256) lmax = fmaxf(lmax, s[c]);
  red[tid] = lmax;
  __syncthreads();
  for (int off = 128; off > 0; off >>= 1) {
    if (tid < off) red[tid] = fmaxf(red[tid], red[tid + off]);
    __syncthreads();
  }
  const float rmax = red[0];
  __syncthreads();

  float lsum = 0.0f;
  for (int c = tid; c < N; c += 256) lsum += __expf(s[c] - rmax);
  red[tid] = lsum;
  __syncthreads();
  for (int off = 128; off > 0; off >>= 1) {
    if (tid < off) red[tid] += red[tid + off];
    __syncthreads();
  }
  const float inv = 1.0f / red[0];

  for (int c = tid; c < N; c += 256)
    p[c] = f32_to_bf16_rne(__expf(s[c] - rmax) * inv);
}

// ---------------------------------------------------------------------------
// Host-side launch
// ---------------------------------------------------------------------------
extern "C" void kernel_launch(void* const* d_in, const int* in_sizes, int n_in,
                              void* d_out, int out_size, void* d_ws, size_t ws_size,
                              hipStream_t stream) {
  constexpr int NT = 4096;  // tokens
  constexpr int D  = 1024;  // feature dim
  const float inv_sqrt_d = 0.03125f;  // 1/sqrt(1024)

  const float* x  = (const float*)d_in[0];
  const float* Wq = (const float*)d_in[1];
  const float* Wk = (const float*)d_in[2];
  const float* Wv = (const float*)d_in[3];
  const float* bq = (const float*)d_in[4];
  const float* bk = (const float*)d_in[5];
  const float* bv = (const float*)d_in[6];
  float* out = (float*)d_out;

  // Workspace layout:
  unsigned short* xb   = (unsigned short*)d_ws;      // x    [NT,D] bf16
  unsigned short* wqtb = xb   + (size_t)NT * D;      // Wq^T [D,D]  bf16
  unsigned short* wktb = wqtb + (size_t)D * D;       // Wk^T
  unsigned short* wvtb = wktb + (size_t)D * D;       // Wv^T
  unsigned short* qb   = wvtb + (size_t)D * D;       // Q [NT,D] bf16
  unsigned short* kb   = qb   + (size_t)NT * D;      // K [NT,D] bf16
  unsigned short* vtb  = kb   + (size_t)NT * D;      // V^T [D,NT] bf16
  float*          sc   = (float*)(vtb + (size_t)NT * D);          // [NT,NT]
  unsigned short* pb   = (unsigned short*)(sc + (size_t)NT * NT); // [NT,NT]

  // 1) downcast x; downcast+transpose weights (so all GEMM B's are [N,K]).
  cvt_f32_to_bf16<<<4096, 256, 0, stream>>>(x, xb, NT * D);
  {
    dim3 g(D / 32, D / 32);
    cvt_transpose_f32_to_bf16<<<g, 256, 0, stream>>>(Wq, wqtb, D);
    cvt_transpose_f32_to_bf16<<<g, 256, 0, stream>>>(Wk, wktb, D);
    cvt_transpose_f32_to_bf16<<<g, 256, 0, stream>>>(Wv, wvtb, D);
  }

  // 2) projections: Q,K normal bf16; V written transposed (V^T [D,NT]).
  {
    dim3 grid(D / BN, NT / BM);
    wmma_gemm<true, true, false><<<grid, 256, 0, stream>>>(
        xb, wqtb, bq, qb, /*K=*/D, D, D, /*ldd=*/D, 1.0f);
    wmma_gemm<true, true, false><<<grid, 256, 0, stream>>>(
        xb, wktb, bk, kb, D, D, D, D, 1.0f);
    wmma_gemm<true, true, true><<<grid, 256, 0, stream>>>(
        xb, wvtb, bv, vtb, D, D, D, /*ldd(T)=*/NT, 1.0f);
  }

  // 3) scores = (Q @ K^T)/sqrt(d): B = K is already [N,K] row-major.
  {
    dim3 grid(NT / BN, NT / BM);
    wmma_gemm<false, false, false><<<grid, 256, 0, stream>>>(
        qb, kb, nullptr, sc, /*K=*/D, D, D, /*ldd=*/NT, inv_sqrt_d);
  }

  // 4) softmax rows -> bf16 P.
  softmax_rows<<<NT, 256, 0, stream>>>(sc, pb, NT);

  // 5) out = P @ V: B = V^T [D rows, NT cols] row-major = [N,K] form.
  {
    dim3 grid(D / BN, NT / BM);
    wmma_gemm<false, false, false><<<grid, 256, 0, stream>>>(
        pb, vtb, nullptr, out, /*K=*/NT, NT, NT, /*ldd=*/D, 1.0f);
  }
}
